// FFN_61400852463649
// MI455X (gfx1250) — 8-bit matrix operand order repaired and checked by reading, compiled, not run on hardware
//
#include <hip/hip_runtime.h>
#include <math.h>

typedef int v8i __attribute__((ext_vector_type(8)));

#define EPS_RMS 1e-8f
#define EPS_Q   1e-5f


__device__ __forceinline__ int q8(float v) {
    float r = rintf(v);
    r = fminf(fmaxf(r, -128.f), 127.f);
    return (int)r;
}

__device__ __forceinline__ v8i mk8(int4 a, int4 b) {
    v8i r;
    r[0] = a.x; r[1] = a.y; r[2] = a.z; r[3] = a.w;
    r[4] = b.x; r[5] = b.y; r[6] = b.z; r[7] = b.w;
    return r;
}

__device__ __forceinline__ int a_vgpr(int kt) {
    return ((kt >> 5) & 1) * 4 + ((kt >> 2) & 3);
}
__device__ __forceinline__ int a_lanehalf(int kt) { return (kt >> 4) & 1; }

__global__ void wq_kernel(const float* __restrict__ w, int N, int K,
                          int Kt, int Nt,
                          int* __restrict__ frag, float* __restrict__ wsOut) {
    __shared__ float red[256];
    __shared__ float s_ws;
    const int tid = threadIdx.x;
    const int total = N * K;

    float acc = 0.f;
    for (int i = tid; i < total; i += 256) acc += fabsf(w[i]);
    red[tid] = acc;
    __syncthreads();
    for (int s = 128; s > 0; s >>= 1) {
        if (tid < s) red[tid] += red[tid + s];
        __syncthreads();
    }
    if (tid == 0) {
        float mean = red[0] / (float)total;
        float wsc = fmaxf(mean, EPS_Q);
        s_ws = wsc;
        *wsOut = wsc;
    }
    __syncthreads();
    const float inv = 1.f / s_ws;

    const int nInts = Kt * Nt * 256;
    for (int idx = tid; idx < nInts; idx += 256) {
        int v    = idx & 7;
        int lane = (idx >> 3) & 31;
        int tn   = idx >> 8;
        int nt   = tn % Nt;
        int t    = tn / Nt;
        int n    = nt * 16 + (lane & 15);
        int kb   = t * 64 + ((v < 4) ? v * 4 : 32 + (v - 4) * 4) + (lane >> 4) * 16;
        unsigned packed = 0;
        #pragma unroll
        for (int j = 0; j < 4; ++j) {
            int k = kb + j;
            int q = 0;
            if (k < K && n < N) {
                float r = rintf(w[(size_t)n * K + k] * inv);
                q = (int)fminf(fmaxf(r, -1.f), 1.f);
            }
            packed |= ((unsigned)(q & 0xff)) << (8 * j);
        }
        frag[idx] = (int)packed;
    }
}

__global__ __launch_bounds__(128)
void bitmlp_kernel(const float* __restrict__ x,
                   const float* __restrict__ s1,
                   const float* __restrict__ s2,
                   const float* __restrict__ s3,
                   const int*  __restrict__ B1,
                   const int*  __restrict__ B2,
                   const int*  __restrict__ B3,
                   const float* __restrict__ wsc,
                   float* __restrict__ out) {
    __shared__ __align__(16) int   Abuf[4 * 13 * 256];
    __shared__ float s1s[784];

    const int tid  = threadIdx.x;
    const int wave = tid >> 5;
    const int lane = tid & 31;
    const int col  = lane & 15;
    const int half = lane >> 4;

    for (int i = tid; i < 784; i += 128) s1s[i] = s1[i];

    const float ws1 = wsc[0], ws2 = wsc[1], ws3 = wsc[2];
    const int tok0 = blockIdx.x * 64 + wave * 16;
    int* Aw = Abuf + wave * 13 * 256;
    __syncthreads();

    float rs1[16];
    #pragma unroll
    for (int r = 0; r < 16; ++r) {
        const float4* xr  = (const float4*)(x + (size_t)(tok0 + r) * 784);
        const float4* s14 = (const float4*)s1s;
        float ss = 0.f, am = 0.f;
        for (int c = lane; c < 196; c += 32) {
            float4 v = xr[c]; float4 s = s14[c];
            ss += v.x * v.x + v.y * v.y + v.z * v.z + v.w * v.w;
            am = fmaxf(am, fmaxf(fmaxf(fabsf(v.x * s.x), fabsf(v.y * s.y)),
                                 fmaxf(fabsf(v.z * s.z), fabsf(v.w * s.w))));
        }
        #pragma unroll
        for (int m = 1; m <= 16; m <<= 1) {
            ss += __shfl_xor(ss, m, 32);
            am  = fmaxf(am, __shfl_xor(am, m, 32));
        }
        float denom = sqrtf(ss * (1.f / 784.f)) + EPS_RMS;
        float amax  = fmaxf(am / denom, EPS_Q);
        float f     = (127.f / amax) / denom;
        rs1[r] = amax * (1.f / 127.f) * ws1;

        for (int c = lane; c < 208; c += 32) {
            int qa = 0, qb = 0, qc = 0, qd = 0;
            if (c < 196) {
                float4 v = xr[c]; float4 s = s14[c];
                qa = q8(v.x * s.x * f); qb = q8(v.y * s.y * f);
                qc = q8(v.z * s.z * f); qd = q8(v.w * s.w * f);
            }
            int packed = (qa & 255) | ((qb & 255) << 8) |
                         ((qc & 255) << 16) | ((qd & 255) << 24);
            int t   = c >> 4;
            int kt4 = (c & 15) * 4;
            int av  = a_vgpr(kt4);
            int lA  = a_lanehalf(kt4) * 16 + r;
            Aw[t * 256 + lA * 8 + av] = packed;
        }
    }
    __syncthreads();

    const v8i vzero = {0, 0, 0, 0, 0, 0, 0, 0};
    v8i acc[8];
    #pragma unroll
    for (int nt = 0; nt < 8; ++nt) acc[nt] = vzero;

    const int4* Aw4 = (const int4*)Aw;
    const int4* B14 = (const int4*)B1;
    #pragma unroll
    for (int t = 0; t < 13; ++t) {
        int4 a0 = Aw4[t * 64 + lane * 2];
        int4 a1 = Aw4[t * 64 + lane * 2 + 1];
        v8i Av = mk8(a0, a1);
        #pragma unroll
        for (int nt = 0; nt < 8; ++nt) {
            int fi = ((t * 8 + nt) * 32 + lane) * 2;
            v8i Bv = mk8(B14[fi], B14[fi + 1]);
            acc[nt] = __builtin_amdgcn_wmma_i32_16x16x64_iu8(
                true, Av, true, Bv, acc[nt], false, false);
        }
    }

    float hv[8][8];
    #pragma unroll
    for (int nt = 0; nt < 8; ++nt)
        #pragma unroll
        for (int i = 0; i < 8; ++i) {
            float rs = half ? rs1[i + 8] : rs1[i];
            hv[nt][i] = fmaxf((float)acc[nt][i] * rs, 0.f);
        }

    float s2r[8];
    #pragma unroll
    for (int nt = 0; nt < 8; ++nt) s2r[nt] = s2[nt * 16 + col];

    float f2[8], rs2[8];
    #pragma unroll
    for (int i = 0; i < 8; ++i) {
        float ss = 0.f, am = 0.f;
        #pragma unroll
        for (int nt = 0; nt < 8; ++nt) {
            float h = hv[nt][i];
            ss += h * h;
            am = fmaxf(am, fabsf(h * s2r[nt]));
        }
        #pragma unroll
        for (int m = 1; m <= 8; m <<= 1) {
            ss += __shfl_xor(ss, m, 32);
            am  = fmaxf(am, __shfl_xor(am, m, 32));
        }
        float denom = sqrtf(ss * (1.f / 128.f)) + EPS_RMS;
        float amax  = fmaxf(am / denom, EPS_Q);
        f2[i]  = (127.f / amax) / denom;
        rs2[i] = amax * (1.f / 127.f) * ws2;
    }
    char* Ab = (char*)Aw;
    #pragma unroll
    for (int nt = 0; nt < 8; ++nt)
        #pragma unroll
        for (int i = 0; i < 8; ++i) {
            int q  = q8(hv[nt][i] * s2r[nt] * f2[i]);
            int k  = nt * 16 + col;
            int t  = k >> 6, kt = k & 63;
            int lA = a_lanehalf(kt) * 16 + (i + half * 8);
            Ab[t * 1024 + lA * 32 + a_vgpr(kt) * 4 + (kt & 3)] = (char)q;
        }
    __syncthreads();

    v8i acc2[4];
    #pragma unroll
    for (int nt = 0; nt < 4; ++nt) acc2[nt] = vzero;
    const int4* B24 = (const int4*)B2;
    #pragma unroll
    for (int t = 0; t < 2; ++t) {
        int4 a0 = Aw4[t * 64 + lane * 2];
        int4 a1 = Aw4[t * 64 + lane * 2 + 1];
        v8i Av = mk8(a0, a1);
        #pragma unroll
        for (int nt = 0; nt < 4; ++nt) {
            int fi = ((t * 4 + nt) * 32 + lane) * 2;
            v8i Bv = mk8(B24[fi], B24[fi + 1]);
            acc2[nt] = __builtin_amdgcn_wmma_i32_16x16x64_iu8(
                true, Av, true, Bv, acc2[nt], false, false);
        }
    }
    float h2[4][8];
    #pragma unroll
    for (int nt = 0; nt < 4; ++nt)
        #pragma unroll
        for (int i = 0; i < 8; ++i)
            h2[nt][i] = fmaxf((float)acc2[nt][i] * rs2[i], 0.f);

    float s3r[4];
    #pragma unroll
    for (int nt = 0; nt < 4; ++nt) s3r[nt] = s3[nt * 16 + col];

    float f3[8], rs3[8];
    #pragma unroll
    for (int i = 0; i < 8; ++i) {
        float ss = 0.f, am = 0.f;
        #pragma unroll
        for (int nt = 0; nt < 4; ++nt) {
            float h = h2[nt][i];
            ss += h * h;
            am = fmaxf(am, fabsf(h * s3r[nt]));
        }
        #pragma unroll
        for (int m = 1; m <= 8; m <<= 1) {
            ss += __shfl_xor(ss, m, 32);
            am  = fmaxf(am, __shfl_xor(am, m, 32));
        }
        float denom = sqrtf(ss * (1.f / 64.f)) + EPS_RMS;
        float amax  = fmaxf(am / denom, EPS_Q);
        f3[i]  = (127.f / amax) / denom;
        rs3[i] = amax * (1.f / 127.f) * ws3;
    }
    #pragma unroll
    for (int nt = 0; nt < 4; ++nt)
        #pragma unroll
        for (int i = 0; i < 8; ++i) {
            int q  = q8(h2[nt][i] * s3r[nt] * f3[i]);
            int kt = nt * 16 + col;
            int lA = a_lanehalf(kt) * 16 + (i + half * 8);
            Ab[lA * 32 + a_vgpr(kt) * 4 + (kt & 3)] = (char)q;
        }
    __syncthreads();

    v8i acc3 = vzero;
    {
        int4 a0 = Aw4[lane * 2];
        int4 a1 = Aw4[lane * 2 + 1];
        v8i Av = mk8(a0, a1);
        const int4* B34 = (const int4*)B3;
        v8i Bv = mk8(B34[lane * 2], B34[lane * 2 + 1]);
        acc3 = __builtin_amdgcn_wmma_i32_16x16x64_iu8(
            true, Av, true, Bv, acc3, false, false);
    }
    #pragma unroll
    for (int i = 0; i < 8; ++i) {
        if (col < 10) {
            int m = i + half * 8;
            out[(size_t)(tok0 + m) * 10 + col] = (float)acc3[i] * rs3[i];
        }
    }
}

extern "C" void kernel_launch(void* const* d_in, const int* in_sizes, int n_in,
                              void* d_out, int out_size, void* d_ws, size_t ws_size,
                              hipStream_t stream) {
    const float* x  = (const float*)d_in[0];
    const float* w1 = (const float*)d_in[1];
    const float* s1 = (const float*)d_in[2];
    const float* w2 = (const float*)d_in[3];
    const float* s2 = (const float*)d_in[4];
    const float* w3 = (const float*)d_in[5];
    const float* s3 = (const float*)d_in[6];
    float* out = (float*)d_out;

    float* wsf = (float*)d_ws;
    int* f1 = (int*)d_ws + 64;
    int* f2 = f1 + 13 * 8 * 256;
    int* f3 = f2 + 2 * 4 * 256;

    wq_kernel<<<1, 256, 0, stream>>>(w1, 128, 784, 13, 8, f1, wsf + 0);
    wq_kernel<<<1, 256, 0, stream>>>(w2,  64, 128,  2, 4, f2, wsf + 1);
    wq_kernel<<<1, 256, 0, stream>>>(w3,  10,  64,  1, 1, f3, wsf + 2);

    bitmlp_kernel<<<65536 / 64, 128, 0, stream>>>(x, s1, s2, s3, f1, f2, f3, wsf, out);
}
